// VFETRT_4080218931373
// MI455X (gfx1250) — matrix operand order repaired and checked by reading, not compiled, not run on hardware
//
#include <hip/hip_runtime.h>
#include <hip/hip_bf16.h>

typedef _Float16 v16h __attribute__((ext_vector_type(16)));
typedef _Float16 v8h  __attribute__((ext_vector_type(8)));
typedef _Float16 v4h  __attribute__((ext_vector_type(4)));
typedef float    v8f  __attribute__((ext_vector_type(8)));

#define NXG 468
#define NYG 468
#define CANVAS (NXG * NYG)

// ---------------------------------------------------------------------------
// Voxel index (NZ == 1, so cz is always 0 and flat = cy*NX + cx)
// ---------------------------------------------------------------------------
__device__ __forceinline__ int voxel_flat(float x, float y, int& cx, int& cy) {
    int ix = (int)floorf((x + 74.88f) / 0.32f);
    int iy = (int)floorf((y + 74.88f) / 0.32f);
    cx = min(max(ix, 0), NXG - 1);
    cy = min(max(iy, 0), NYG - 1);
    return cy * NXG + cx;
}

// ---------------------------------------------------------------------------
// WMMA fragment loaders (wave32, 16x16x32 f16 shapes)
// A (16xK f16): lane holds row (lane&15); halves {kb+hi*8..+7, kb+16+hi*8..+7}
// B (Kx16 f16): lane holds col (lane&15); halves {kb+hi*8..+7, kb+16+hi*8..+7}
// ---------------------------------------------------------------------------
__device__ __forceinline__ v16h frag_a(const _Float16* rowbase, int hi, int kb) {
    v8h lo = *(const v8h*)(rowbase + kb + hi * 8);
    v8h hv = *(const v8h*)(rowbase + kb + 16 + hi * 8);
    v16h r;
#pragma unroll
    for (int i = 0; i < 8; ++i) { r[i] = lo[i]; r[i + 8] = hv[i]; }
    return r;
}

__device__ __forceinline__ v16h frag_b(const _Float16* colbase, int hi, int kb) {
    v8h lo = *(const v8h*)(colbase + kb + hi * 8);
    v8h hv = *(const v8h*)(colbase + kb + 16 + hi * 8);
    v16h r;
#pragma unroll
    for (int i = 0; i < 8; ++i) { r[i] = lo[i]; r[i + 8] = hv[i]; }
    return r;
}

// ---------------------------------------------------------------------------
// K0: zero a float buffer, vectorized (n must be a multiple of 4 — it is)
// ---------------------------------------------------------------------------
__global__ void k_zero4(float4* __restrict__ p, int n4) {
    int i = blockIdx.x * blockDim.x + threadIdx.x;
    int stride = gridDim.x * blockDim.x;
    const float4 z = make_float4(0.f, 0.f, 0.f, 0.f);
    for (; i < n4; i += stride) p[i] = z;
}

// ---------------------------------------------------------------------------
// K1: per-point scatter-add of count and sum(xyz)
// ---------------------------------------------------------------------------
__global__ void k_scatter(const float* __restrict__ pts, int N,
                          float* __restrict__ cnt, float* __restrict__ vsum) {
    int p = blockIdx.x * blockDim.x + threadIdx.x;
    if (p >= N) return;
    const float* P = pts + (size_t)p * 5;
    float x = P[0], y = P[1], z = P[2];
    int cx, cy;
    int flat = voxel_flat(x, y, cx, cy);
    atomicAdd(&cnt[flat], 1.0f);
    atomicAdd(&vsum[flat * 3 + 0], x);
    atomicAdd(&vsum[flat * 3 + 1], y);
    atomicAdd(&vsum[flat * 3 + 2], z);
}

// ---------------------------------------------------------------------------
// Shared helper: build the 11-dim feature row (f16, zero-padded to 32) for one
// point into an LDS A-tile, and record its flat voxel index.
// ---------------------------------------------------------------------------
__device__ __forceinline__ void build_features(const float* __restrict__ pts,
                                               const float* __restrict__ cnt,
                                               const float* __restrict__ vsum,
                                               int p, int N,
                                               _Float16* row, int* flat_slot) {
    float f[11];
    int flat = 0;
    if (p < N) {
        const float* P = pts + (size_t)p * 5;
        float x = P[0], y = P[1], z = P[2];
        int cx, cy;
        flat = voxel_flat(x, y, cx, cy);
        float c   = cnt[flat];
        float inv = 1.0f / fmaxf(c, 1.0f);
        float mx = vsum[flat * 3 + 0] * inv;
        float my = vsum[flat * 3 + 1] * inv;
        float mz = vsum[flat * 3 + 2] * inv;
        f[0] = x; f[1] = y; f[2] = z; f[3] = P[3]; f[4] = P[4];
        f[5] = x - mx; f[6] = y - my; f[7] = z - mz;
        f[8] = x - ((float)cx * 0.32f - 74.72f);   // OFFSET = VOXEL/2 + PC_MIN
        f[9] = y - ((float)cy * 0.32f - 74.72f);
        f[10] = z - 1.0f;                          // cz==0 -> center z = 1.0
    } else {
#pragma unroll
        for (int j = 0; j < 11; ++j) f[j] = 0.0f;
    }
    *flat_slot = flat;
#pragma unroll
    for (int j = 0; j < 11; ++j) row[j] = (_Float16)f[j];
#pragma unroll
    for (int j = 11; j < 32; ++j) row[j] = (_Float16)0.0f;
}

// ---------------------------------------------------------------------------
// Scatter epilogue: 4 N-tiles of D (relu >= 0 -> integer-bit atomicMax is
// exact with 0-initialized buffers).  m-outer / t-inner so the 4 atomics per
// row share one base address with constant byte offsets (0/64/128/192).
// ---------------------------------------------------------------------------
__device__ __forceinline__ void scatter_max(const v8f* dt, const int* fl,
                                            float* __restrict__ dst,
                                            int lo, int hi, int tileBase, int N) {
    if (tileBase + 16 <= N) {                     // wave-uniform fast path
#pragma unroll
        for (int r = 0; r < 8; ++r) {
            int* bp = (int*)(dst + (size_t)fl[r] * 64 + lo);
#pragma unroll
            for (int t = 0; t < 4; ++t)
                atomicMax(bp + t * 16, __float_as_int(fmaxf(dt[t][r], 0.0f)));
        }
    } else {                                      // ragged last tile only
#pragma unroll
        for (int r = 0; r < 8; ++r) {
            if (tileBase + r + 8 * hi < N) {
                int* bp = (int*)(dst + (size_t)fl[r] * 64 + lo);
#pragma unroll
                for (int t = 0; t < 4; ++t)
                    atomicMax(bp + t * 16, __float_as_int(fmaxf(dt[t][r], 0.0f)));
            }
        }
    }
}

// ---------------------------------------------------------------------------
// K2 (phase 1): features -> pf1 = relu(feat @ W1) via WMMA -> segment-max(v1)
// 8 waves/block, one 16-point tile per wave.
// ---------------------------------------------------------------------------
__global__ __launch_bounds__(256) void k_phase1(
    const float* __restrict__ pts, const float* __restrict__ W1, int N,
    const float* __restrict__ cnt, const float* __restrict__ vsum,
    float* __restrict__ v1max) {
    __shared__ __align__(16) _Float16 Bt1[64 * 32];      // W1^T, K padded 11->32
    __shared__ __align__(16) _Float16 A1[8][16 * 32];
    __shared__ __align__(16) int flatS[8][16];

    const int tid = threadIdx.x;
    for (int i = tid; i < 64 * 32; i += 256) {
        int n = i >> 5, k = i & 31;
        Bt1[n * 32 + k] = (k < 11) ? (_Float16)W1[k * 64 + n] : (_Float16)0.0f;
    }
    const int wave = tid >> 5, lane = tid & 31;
    const int lo = lane & 15, hi = lane >> 4;
    const int tileBase = (blockIdx.x * 8 + wave) * 16;

    if (lane < 16)
        build_features(pts, cnt, vsum, tileBase + lane, N,
                       &A1[wave][lane * 32], &flatS[wave][lane]);
    __syncthreads();

    v16h a = frag_a(&A1[wave][lo * 32], hi, 0);
    v8f dt[4];
#pragma unroll
    for (int t = 0; t < 4; ++t) {
        v16h b = frag_b(&Bt1[(t * 16 + lo) * 32], hi, 0);
        v8f c = {};
        dt[t] = __builtin_amdgcn_wmma_f32_16x16x32_f16(false, a, false, b,
                                                       (short)0, c, false, false);
    }
    int fl[8];                                    // 8 consecutive ints -> b128s
#pragma unroll
    for (int r = 0; r < 8; ++r) fl[r] = flatS[wave][r + 8 * hi];

    scatter_max(dt, fl, v1max, lo, hi, tileBase, N);
}

// ---------------------------------------------------------------------------
// K3 (phase 2): recompute pf1 (cheaper than spilling 102 MB), append v1[flat],
// pf2 = relu(feat2 @ W2) via WMMA -> segment-max into d_out.
// ---------------------------------------------------------------------------
__global__ __launch_bounds__(256) void k_phase2(
    const float* __restrict__ pts, const float* __restrict__ W1,
    const float* __restrict__ W2, int N,
    const float* __restrict__ cnt, const float* __restrict__ vsum,
    const float* __restrict__ v1max, float* __restrict__ out) {
    __shared__ __align__(16) _Float16 Bt1[64 * 32];      //  4 KB
    __shared__ __align__(16) _Float16 Bt2[64 * 128];     // 16 KB, W2^T
    __shared__ __align__(16) _Float16 A1[8][16 * 32];    //  8 KB
    __shared__ __align__(16) _Float16 A2[8][16 * 128];   // 32 KB, [pf1 | v1]
    __shared__ __align__(16) int flatS[8][16];

    const int tid = threadIdx.x;
    for (int i = tid; i < 64 * 32; i += 256) {
        int n = i >> 5, k = i & 31;
        Bt1[n * 32 + k] = (k < 11) ? (_Float16)W1[k * 64 + n] : (_Float16)0.0f;
    }
    for (int i = tid; i < 64 * 128; i += 256) {
        int n = i >> 7, k = i & 127;
        Bt2[n * 128 + k] = (_Float16)W2[k * 64 + n];
    }
    const int wave = tid >> 5, lane = tid & 31;
    const int lo = lane & 15, hi = lane >> 4;
    const int tileBase = (blockIdx.x * 8 + wave) * 16;

    if (lane < 16)
        build_features(pts, cnt, vsum, tileBase + lane, N,
                       &A1[wave][lane * 32], &flatS[wave][lane]);
    __syncthreads();

    // GEMM1: pf1 tile -> A2 columns [0,64)
    v16h a1 = frag_a(&A1[wave][lo * 32], hi, 0);
    v8f d1[4];
#pragma unroll
    for (int t = 0; t < 4; ++t) {
        v16h b = frag_b(&Bt1[(t * 16 + lo) * 32], hi, 0);
        v8f c = {};
        d1[t] = __builtin_amdgcn_wmma_f32_16x16x32_f16(false, a1, false, b,
                                                       (short)0, c, false, false);
    }
#pragma unroll
    for (int r = 0; r < 8; ++r) {
        _Float16* rowp = &A2[wave][(r + 8 * hi) * 128 + lo];
#pragma unroll
        for (int t = 0; t < 4; ++t)
            rowp[t * 16] = (_Float16)fmaxf(d1[t][r], 0.0f);
    }
    // v1[flat] -> A2 columns [64,128): float4 gather, packed v4h stores
    for (int idx = lane; idx < 16 * 16; idx += 32) {
        int m = idx >> 4, q = idx & 15;
        float4 v = *(const float4*)(v1max + (size_t)flatS[wave][m] * 64 + q * 4);
        v4h h;
        h[0] = (_Float16)v.x; h[1] = (_Float16)v.y;
        h[2] = (_Float16)v.z; h[3] = (_Float16)v.w;
        *(v4h*)&A2[wave][m * 128 + 64 + q * 4] = h;
    }
    __syncthreads();

    // GEMM2: 4 N-tiles x 4 K-steps (K = 128)
    v16h a2[4];
#pragma unroll
    for (int ks = 0; ks < 4; ++ks)
        a2[ks] = frag_a(&A2[wave][lo * 128], hi, ks * 32);
    v8f d2[4];
#pragma unroll
    for (int t = 0; t < 4; ++t) {
        v8f acc = {};
#pragma unroll
        for (int ks = 0; ks < 4; ++ks) {
            v16h b = frag_b(&Bt2[(t * 16 + lo) * 128], hi, ks * 32);
            acc = __builtin_amdgcn_wmma_f32_16x16x32_f16(false, a2[ks], false, b,
                                                         (short)0, acc, false, false);
        }
        d2[t] = acc;
    }
    int fl[8];
#pragma unroll
    for (int r = 0; r < 8; ++r) fl[r] = flatS[wave][r + 8 * hi];

    scatter_max(d2, fl, out, lo, hi, tileBase, N);
}

// ---------------------------------------------------------------------------
// Launch
// ---------------------------------------------------------------------------
extern "C" void kernel_launch(void* const* d_in, const int* in_sizes, int n_in,
                              void* d_out, int out_size, void* d_ws, size_t ws_size,
                              hipStream_t stream) {
    const float* pts = (const float*)d_in[0];   // (N, 5)
    const float* W1  = (const float*)d_in[1];   // (11, 64)
    const float* W2  = (const float*)d_in[2];   // (128, 64)
    const int N = in_sizes[0] / 5;
    float* out = (float*)d_out;                 // (CANVAS, 64)

    // workspace layout: cnt | vsum(3) | v1max(64), contiguous floats
    float* cnt   = (float*)d_ws;
    float* vsum  = cnt + CANVAS;
    float* v1max = vsum + 3 * CANVAS;

    k_zero4<<<2048, 256, 0, stream>>>((float4*)cnt, 68 * CANVAS / 4);
    k_zero4<<<2048, 256, 0, stream>>>((float4*)out, 64 * CANVAS / 4);

    k_scatter<<<(N + 255) / 256, 256, 0, stream>>>(pts, N, cnt, vsum);

    const int blocks = (N + 127) / 128;   // 128 points per block (8 waves x 16)
    k_phase1<<<blocks, 256, 0, stream>>>(pts, W1, N, cnt, vsum, v1max);
    k_phase2<<<blocks, 256, 0, stream>>>(pts, W1, W2, N, cnt, vsum, v1max, out);
}
